// DynamicSparselyGatedMixtureOfExperts_13073880449223
// MI455X (gfx1250) — matrix operand order repaired and checked by reading, not compiled, not run on hardware
//
#include <hip/hip_runtime.h>
#include <hip/hip_bf16.h>

#define N_TOK 4096
#define DIM   1024
#define NEXP  8
#define MT    64      // M tile (tokens per workgroup) -> 4x B reuse per wave
#define KC    128     // K chunk staged in LDS
#define NCHUNK (DIM / KC)

typedef __attribute__((ext_vector_type(16))) __bf16 v16bf;
typedef __attribute__((ext_vector_type(8)))  __bf16 v8bf;
typedef __attribute__((ext_vector_type(4)))  __bf16 v4bf;
typedef __attribute__((ext_vector_type(8)))  float  v8f;

// ---------------------------------------------------------------- zero init
__global__ void zero_kernel(float* __restrict__ out, int n, int* __restrict__ counts) {
    int i = blockIdx.x * blockDim.x + threadIdx.x;
    if (i < n) out[i] = 0.0f;
    if (i < NEXP) counts[i] = 0;
}

// ---------------------------------------------------------------- gating
__global__ __launch_bounds__(256)
void moe_gating(const float* __restrict__ x, const float* __restrict__ noise,
                const float* __restrict__ Wg, const float* __restrict__ Wn,
                int* __restrict__ counts, int* __restrict__ tok_list,
                float* __restrict__ gate_list) {
    const int lane = threadIdx.x & 31;
    const int wave = threadIdx.x >> 5;
    const int n = blockIdx.x * 8 + wave;
    if (n >= N_TOK) return;

    float ag[NEXP], an[NEXP];
#pragma unroll
    for (int e = 0; e < NEXP; ++e) { ag[e] = 0.f; an[e] = 0.f; }

    const float* xr = x + (size_t)n * DIM;
    for (int d = lane; d < DIM; d += 32) {
        float xv = xr[d];
        const float4* wg4 = reinterpret_cast<const float4*>(Wg + (size_t)d * NEXP);
        const float4* wn4 = reinterpret_cast<const float4*>(Wn + (size_t)d * NEXP);
        float4 g0 = wg4[0], g1 = wg4[1];
        float4 n0 = wn4[0], n1 = wn4[1];
        ag[0] = fmaf(xv, g0.x, ag[0]); ag[1] = fmaf(xv, g0.y, ag[1]);
        ag[2] = fmaf(xv, g0.z, ag[2]); ag[3] = fmaf(xv, g0.w, ag[3]);
        ag[4] = fmaf(xv, g1.x, ag[4]); ag[5] = fmaf(xv, g1.y, ag[5]);
        ag[6] = fmaf(xv, g1.z, ag[6]); ag[7] = fmaf(xv, g1.w, ag[7]);
        an[0] = fmaf(xv, n0.x, an[0]); an[1] = fmaf(xv, n0.y, an[1]);
        an[2] = fmaf(xv, n0.z, an[2]); an[3] = fmaf(xv, n0.w, an[3]);
        an[4] = fmaf(xv, n1.x, an[4]); an[5] = fmaf(xv, n1.y, an[5]);
        an[6] = fmaf(xv, n1.z, an[6]); an[7] = fmaf(xv, n1.w, an[7]);
    }
#pragma unroll
    for (int e = 0; e < NEXP; ++e) {
#pragma unroll
        for (int off = 16; off > 0; off >>= 1) {
            ag[e] += __shfl_down(ag[e], off, 32);
            an[e] += __shfl_down(an[e], off, 32);
        }
    }

    if (lane == 0) {
        float H[NEXP];
#pragma unroll
        for (int e = 0; e < NEXP; ++e) {
            float v  = an[e];
            float sp = (v > 20.f) ? v : log1pf(expf(v));   // stable softplus
            H[e] = ag[e] + noise[e] * sp;
        }
        int i0 = 0;
#pragma unroll
        for (int e = 1; e < NEXP; ++e) if (H[e] > H[i0]) i0 = e;
        int i1 = (i0 == 0) ? 1 : 0;
#pragma unroll
        for (int e = 0; e < NEXP; ++e) if (e != i1 && e != i0 && H[e] > H[i1]) i1 = e;
        float m  = fmaxf(H[i0], H[i1]);
        float e0 = expf(H[i0] - m), e1 = expf(H[i1] - m);
        float inv = 1.0f / (e0 + e1);

        int s0 = atomicAdd(&counts[i0], 1);
        tok_list [i0 * N_TOK + s0] = n;
        gate_list[i0 * N_TOK + s0] = e0 * inv;
        int s1 = atomicAdd(&counts[i1], 1);
        tok_list [i1 * N_TOK + s1] = n;
        gate_list[i1 * N_TOK + s1] = e1 * inv;
    }
}

// ---------------------------------------------------------------- expert GEMM
// Block = 8 wave32 = 64(M) x 128(N) tile of one expert's token group.
// Double-buffered LDS A (bf16); per-thread A-row pointers hoisted to registers
// so next-chunk global loads overlap the current chunk's 16 WMMAs.
__global__ __launch_bounds__(256)
void moe_expert_gemm(const float* __restrict__ x, const float* __restrict__ We,
                     const float* __restrict__ be, const int* __restrict__ counts,
                     const int* __restrict__ tok_list, const float* __restrict__ gate_list,
                     float* __restrict__ out) {
    const int e   = blockIdx.y;
    const int mt  = blockIdx.z;
    const int cnt = counts[e];
    if (mt * MT >= cnt) return;

    __shared__ __bf16 As[2][MT][KC];   // 2 x 16 KB double-buffered A chunk
    __shared__ int    s_tok[MT];
    __shared__ float  s_gate[MT];

    const int tid  = threadIdx.x;
    const int lane = tid & 31;
    const int wave = tid >> 5;

    if (tid < MT) {
        int idx = mt * MT + tid;
        if (idx < cnt) {
            s_tok [tid] = tok_list [e * N_TOK + idx];
            s_gate[tid] = gate_list[e * N_TOK + idx];
        } else {
            s_tok[tid] = 0;
            s_gate[tid] = 0.0f;      // tail rows contribute exactly 0
        }
    }
    __syncthreads();

    // Hoisted per-thread A staging pointers: this thread always stages
    // rows r = wave + 8*it at column block kv = lane*4 (float4 granularity).
    const int kv = lane * 4;
    const float* ap[8];
#pragma unroll
    for (int it = 0; it < 8; ++it)
        ap[it] = x + (size_t)s_tok[wave + it * 8] * DIM + kv;

    const int h0 = blockIdx.x * 128 + wave * 16;
    const float* Wbase = We + (size_t)e * DIM * DIM;   // We[e][k][h], h contiguous

    v8f acc[4] = {{}, {}, {}, {}};
    float4 fbuf[8];

    // prologue: load + store chunk 0 into buffer 0
#pragma unroll
    for (int it = 0; it < 8; ++it)
        fbuf[it] = *reinterpret_cast<const float4*>(ap[it]);
#pragma unroll
    for (int it = 0; it < 8; ++it) {
        v4bf p;
        p[0] = (__bf16)fbuf[it].x; p[1] = (__bf16)fbuf[it].y;
        p[2] = (__bf16)fbuf[it].z; p[3] = (__bf16)fbuf[it].w;
        *reinterpret_cast<v4bf*>(&As[0][wave + it * 8][kv]) = p;
    }
    __syncthreads();

#pragma unroll
    for (int c = 0; c < NCHUNK; ++c) {
        const int kc  = c * KC;
        const int buf = c & 1;

        // issue next chunk's global loads early (latency hides under WMMAs)
        if (c + 1 < NCHUNK) {
#pragma unroll
            for (int it = 0; it < 8; ++it)
                fbuf[it] = *reinterpret_cast<const float4*>(ap[it] + kc + KC);
        }

#pragma unroll
        for (int kk = 0; kk < KC; kk += 32) {
            // B fragment: lane l holds col h0+(l&15); elem i -> K = kc+kk+8*(l>>4)+(i<8 ? i : i+8) (bf16)
            const float* brow = Wbase + (size_t)(kc + kk + (lane >> 4) * 8) * DIM + h0 + (lane & 15);
            v16bf b;
#pragma unroll
            for (int i = 0; i < 16; ++i) {
                const int kb = (i < 8) ? i : i + 8;
                b[i] = (__bf16)brow[(size_t)kb * DIM];
            }
            if (c + 1 < NCHUNK)   // prefetch next K-chunk's B row (global_prefetch_b8)
                __builtin_prefetch(brow + (size_t)KC * DIM, 0, 3);

            // A fragments per ISA 16-bit A layout:
            // lane<16: K {0..7,16..23}; lane>=16: K {8..15,24..31} of row (lane&15)
            const int r0  = lane & 15;
            const int hi  = lane >> 4;
            const int ka0 = kk + hi * 8;
            const int ka1 = kk + 16 + hi * 8;

            // materialize all 4 A fragments first (one DS burst, distinct regs)
            v16bf afrag[4];
#pragma unroll
            for (int s = 0; s < 4; ++s) {
                const int r = s * 16 + r0;
                v8bf a_lo = *reinterpret_cast<const v8bf*>(&As[buf][r][ka0]);
                v8bf a_hi = *reinterpret_cast<const v8bf*>(&As[buf][r][ka1]);
#pragma unroll
                for (int i = 0; i < 8; ++i) {
                    afrag[s][i]     = a_lo[i];
                    afrag[s][i + 8] = a_hi[i];
                }
            }
            // then the WMMA quartet sharing this B fragment
#pragma unroll
            for (int s = 0; s < 4; ++s) {
                acc[s] = __builtin_amdgcn_wmma_f32_16x16x32_bf16(
                    /*neg_a=*/false, afrag[s], /*neg_b=*/false, b,
                    /*c_mod=*/(short)0, acc[s], /*reuse_a=*/false, /*reuse_b=*/false);
            }
        }

        // convert + store next chunk into the other buffer (its old readers
        // finished at the barrier that ended iteration c-1)
        if (c + 1 < NCHUNK) {
#pragma unroll
            for (int it = 0; it < 8; ++it) {
                v4bf p;
                p[0] = (__bf16)fbuf[it].x; p[1] = (__bf16)fbuf[it].y;
                p[2] = (__bf16)fbuf[it].z; p[3] = (__bf16)fbuf[it].w;
                *reinterpret_cast<v4bf*>(&As[buf ^ 1][wave + it * 8][kv]) = p;
            }
            __syncthreads();
        }
    }

    // Epilogue. C/D layout: VGPR j -> M = j + 8*(lane>=16), N = lane&15.
    const int col  = lane & 15;
    const int half = lane >> 4;
    const float bias = be[(size_t)e * DIM + h0 + col];
#pragma unroll
    for (int s = 0; s < 4; ++s) {
#pragma unroll
        for (int j = 0; j < 8; ++j) {
            int   m = s * 16 + j + half * 8;
            float g = s_gate[m];
            float v = g * (acc[s][j] + bias);
            atomicAdd(&out[(size_t)s_tok[m] * DIM + h0 + col], v);
        }
    }
}

// ---------------------------------------------------------------- launch
extern "C" void kernel_launch(void* const* d_in, const int* in_sizes, int n_in,
                              void* d_out, int out_size, void* d_ws, size_t ws_size,
                              hipStream_t stream) {
    const float* x     = (const float*)d_in[0];
    const float* noise = (const float*)d_in[1];
    const float* Wg    = (const float*)d_in[2];
    const float* Wn    = (const float*)d_in[3];
    const float* We    = (const float*)d_in[4];
    const float* be    = (const float*)d_in[5];
    float* out = (float*)d_out;

    // workspace layout (~260 KB)
    int*   counts    = (int*)d_ws;
    int*   tok_list  = (int*)((char*)d_ws + 256);
    float* gate_list = (float*)((char*)d_ws + 256 + (size_t)NEXP * N_TOK * sizeof(int));

    const int out_elems = N_TOK * DIM;
    zero_kernel<<<(out_elems + 255) / 256, 256, 0, stream>>>(out, out_elems, counts);

    moe_gating<<<N_TOK / 8, 256, 0, stream>>>(x, noise, Wg, Wn,
                                              counts, tok_list, gate_list);

    dim3 grid(DIM / 128, NEXP, N_TOK / MT);   // col-tiles x experts x max M-tiles
    moe_expert_gemm<<<grid, 256, 0, stream>>>(x, We, be, counts,
                                              tok_list, gate_list, out);
}
